// LigerLinearCrossEntropy_80410377715905
// MI455X (gfx1250) — compile-verified, matrix operand order checked by reading, not run on hardware
//
#include <hip/hip_runtime.h>

// ---------------------------------------------------------------------------
// Fused Linear Cross-Entropy for MI455X (gfx1250, wave32, WMMA bf16 + TDM)
//   loss = mean_{t!=-100} [ logsumexp(x_n . W^T + b) - (x_n . W_t + b_t) ]
// Never materializes the [8192 x 32000] logits. Memory-bound: bf16 weight
// stream (~131 MB) dominates. Matmul: v_wmma_f32_16x16x32_bf16; LDS panels
// staged by the Tensor Data Mover (tensor_load_to_lds), double-buffered.
// ---------------------------------------------------------------------------

typedef unsigned short u16;
typedef unsigned int   u32;
typedef unsigned long long u64;
typedef __attribute__((ext_vector_type(16))) __bf16 v16bf;
typedef __attribute__((ext_vector_type(8)))  float  v8f;
typedef __attribute__((ext_vector_type(4)))  unsigned int v4u;
typedef __attribute__((ext_vector_type(8)))  int v8i;
typedef __attribute__((ext_vector_type(4)))  int v4i;

#define N_ROWS   8192      // 4 * 2048 tokens
#define EMB      2048
#define VOCAB    32000
#define M_BLK    64        // rows per block
#define V_CHUNK  1280      // cols per block (25 chunks)
#define CB_COLS  128       // cols per column-block iteration (10 iters)
#define K_PANEL  64        // K halves staged per panel (32 panels)
#define N_PANEL  (EMB / K_PANEL)
#define LDS_STR  72        // padded row stride (halves): 64 + 8 pad
#define N_PART   100       // partials per row: 25 chunks * 4 column waves

// LDS layout (halves), double-buffered panels; smem assumed at LDS offset 0
#define A_HALVES (M_BLK * LDS_STR)     // 4608
#define B_HALVES (CB_COLS * LDS_STR)   // 9216
#define SMEM_HALVES (2 * A_HALVES + 2 * B_HALVES)   // 27648 (55296 B)

// workspace layout (bytes)
#define OFF_WBF  0ull                               // 32000*2048*2 = 131072000
#define OFF_HBF  131072000ull                       //  8192*2048*2 =  33554432
#define OFF_TLG  164626432ull                       //  8192*4
#define OFF_PRT  164659200ull                       //  8192*100*8
#define OFF_BSM  171212800ull                       //  32*8

__device__ __forceinline__ u16 f2bf_rne(float f) {
  u32 u = __float_as_uint(f);
  u32 r = u + 0x7FFFu + ((u >> 16) & 1u);
  return (u16)(r >> 16);
}

// ---- kernel 0a: f32 -> bf16 bulk convert -----------------------------------
__global__ __launch_bounds__(256) void flce_convert(const float* __restrict__ src,
                                                    u16* __restrict__ dst,
                                                    size_t n8) {
  size_t i = (size_t)blockIdx.x * blockDim.x + threadIdx.x;
  size_t stride = (size_t)gridDim.x * blockDim.x;
  const float4* s4 = reinterpret_cast<const float4*>(src);
  uint4* d4 = reinterpret_cast<uint4*>(dst);
  for (; i < n8; i += stride) {
    float4 a = s4[2 * i], b = s4[2 * i + 1];
    uint4 o;
    o.x = (u32)f2bf_rne(a.x) | ((u32)f2bf_rne(a.y) << 16);
    o.y = (u32)f2bf_rne(a.z) | ((u32)f2bf_rne(a.w) << 16);
    o.z = (u32)f2bf_rne(b.x) | ((u32)f2bf_rne(b.y) << 16);
    o.w = (u32)f2bf_rne(b.z) | ((u32)f2bf_rne(b.w) << 16);
    d4[i] = o;
  }
}

// ---- kernel 0b: zero the target-logit buffer -------------------------------
__global__ void flce_zero(float* __restrict__ p, int n) {
  int i = blockIdx.x * blockDim.x + threadIdx.x;
  if (i < n) p[i] = 0.0f;
}

// ---- TDM: issue a 2-D tile load (rows x 64 halves) into LDS ----------------
// data_size = 4B units; tile_dim0 = 32 dw (=64 halves); row stride 1024 dw
// (=2048 halves); LDS padding: +4 dw after every 32 dw -> LDS_STR = 72 halves.
#if defined(__has_builtin)
#if __has_builtin(__builtin_amdgcn_tensor_load_to_lds)
#define FLCE_HAVE_TDM 1
#endif
#endif

#ifdef FLCE_HAVE_TDM
__device__ __forceinline__ void tdm_load_panel(u64 gaddr, u32 lds_byte,
                                               int rows, int tensor_rows) {
  v4u g0;
  g0.x = 1u;                                              // count=1, user load
  g0.y = lds_byte;                                        // lds_addr
  g0.z = (u32)(gaddr & 0xFFFFFFFFull);                    // global_addr[31:0]
  g0.w = (u32)((gaddr >> 32) & 0x1FFFFFFull) | (2u << 30);// addr[56:32], type=2
  v8i g1;
  g1[0] = (2 << 16)        // data_size = 4 bytes
        | (1 << 20)        // pad_enable
        | (4 << 22)        // pad_interval: 32 DWORDs (one 64-half row)
        | (3 << 25);       // pad_amount: 4 DWORDs (8-half pad)
  g1[1] = (1024 << 16);                        // tensor_dim0[15:0] = 1024 dw
  g1[2] = ((tensor_rows & 0xFFFF) << 16);      // dim0 hi=0 | tensor_dim1 lo
  g1[3] = ((tensor_rows >> 16) & 0xFFFF)       // tensor_dim1 hi
        | (32 << 16);                          // tile_dim0 = 32 dw
  g1[4] = rows & 0xFFFF;                       // tile_dim1 (tile_dim2=0)
  g1[5] = 1024;                                // tensor_dim0_stride lo = 1024 dw
  g1[6] = 0;
  g1[7] = 0;
  v4i gz4 = {0, 0, 0, 0};                      // groups 2/3 unused (2-D tile)
  v8i gz8 = {0, 0, 0, 0, 0, 0, 0, 0};          // unused trailing group
  __builtin_amdgcn_tensor_load_to_lds(g0, g1, gz4, gz4, gz8, 0);
}
#endif

// ---- CDNA5 bf16 A/B fragment load from LDS ---------------------------------
// lane r (=lane&15) holds matrix row r; lane half h (=lane>>4) selects K+8.
// VGPRs 0..3 <- K = koff..koff+7 ; VGPRs 4..7 <- K = koff+16..koff+23.
__device__ __forceinline__ v16bf ld_frag(const u16* rowbase, int koff) {
  union { uint4 q[2]; v16bf v; } f;
  f.q[0] = *reinterpret_cast<const uint4*>(rowbase + koff);
  f.q[1] = *reinterpret_cast<const uint4*>(rowbase + koff + 16);
  return f.v;
}

// ---- kernel 1: streaming GEMM + online logsumexp partials ------------------
__global__ __launch_bounds__(256) void flce_gemm(
    const u16* __restrict__ hbf,      // [8192][2048] bf16
    const u16* __restrict__ wbf,      // [32000][2048] bf16
    const float* __restrict__ bias,   // [32000]
    const int* __restrict__ targets,  // [8192]
    float* __restrict__ tlogit,       // [8192]
    float2* __restrict__ partials) {  // [8192][100] (m, s)
  __shared__ __align__(16) u16 smem[SMEM_HALVES];

#ifdef FLCE_HAVE_TDM
  // The TDM engine writes smem behind the compiler's back (the LDS address in
  // the D# descriptor is an opaque integer). Escape the pointer so LLVM must
  // treat smem as externally written; otherwise loads fold to undef.
  {
    u16* sp = smem;
    asm volatile("" : "+v"(sp) : : "memory");
  }
#endif

  const int tid  = threadIdx.x;
  const int lane = tid & 31;
  const int w    = tid >> 5;       // wave 0..7
  const int wm   = w & 1;          // 2 waves along M (32 rows each)
  const int wn   = w >> 1;         // 4 waves along N (32 cols each)
  const int h    = lane >> 4;      // half-wave
  const int ln   = lane & 15;

  const int m0  = blockIdx.x * M_BLK;
  const int vch = blockIdx.y;
  const int v0  = vch * V_CHUNK;

  const u64 hbase = (u64)(uintptr_t)hbf;
  const u64 wbase = (u64)(uintptr_t)wbf;

  // this thread's 16 rows (one per (mt,j) at its half h)
  int tv[2][8];
#pragma unroll
  for (int mt = 0; mt < 2; ++mt)
#pragma unroll
    for (int j = 0; j < 8; ++j)
      tv[mt][j] = targets[m0 + 32 * wm + 16 * mt + j + 8 * h];

  float m_run[2][8], s_run[2][8];
#pragma unroll
  for (int mt = 0; mt < 2; ++mt)
#pragma unroll
    for (int j = 0; j < 8; ++j) { m_run[mt][j] = -__builtin_inff(); s_run[mt][j] = 0.0f; }

  v8f zero;
#pragma unroll
  for (int i = 0; i < 8; ++i) zero[i] = 0.0f;

  for (int cb = 0; cb < V_CHUNK / CB_COLS; ++cb) {
    const int vb = v0 + cb * CB_COLS;
    v8f acc[2][2];
    acc[0][0] = zero; acc[0][1] = zero; acc[1][0] = zero; acc[1][1] = zero;

#ifdef FLCE_HAVE_TDM
    // prologue: wave 0 DMAs panel 0 into buffer 0 (A then B)
    if (tid < 32) {
      tdm_load_panel(hbase + ((u64)m0 * EMB) * 2, 0u, M_BLK, N_ROWS);
      tdm_load_panel(wbase + ((u64)vb * EMB) * 2, 2u * (2 * A_HALVES),
                     CB_COLS, VOCAB);
    }
#endif

    for (int kp = 0; kp < N_PANEL; ++kp) {
      const int pa = kp & 1;
#ifdef FLCE_HAVE_TDM
      if (tid < 32) {
        if (kp + 1 < N_PANEL) {       // DMA next panel into the other buffer
          const int pn = (kp + 1) & 1;
          const u64 koff = (u64)(kp + 1) * K_PANEL * 2;
          tdm_load_panel(hbase + ((u64)m0 * EMB) * 2 + koff,
                         (u32)(pn * 2 * A_HALVES), M_BLK, N_ROWS);
          tdm_load_panel(wbase + ((u64)vb * EMB) * 2 + koff,
                         (u32)(2 * (2 * A_HALVES) + pn * 2 * B_HALVES),
                         CB_COLS, VOCAB);
          __builtin_amdgcn_s_wait_tensorcnt((short)2);  // panel kp complete
        } else {
          __builtin_amdgcn_s_wait_tensorcnt((short)0);
        }
      }
      __syncthreads();
#else
      // fallback: manual staging (global_load_b128 -> ds_store_b128)
      {
        const int ar = tid >> 2, aq = tid & 3;
        const int bc = tid >> 1, bq = tid & 1;
        u16* As = smem + pa * A_HALVES;
        u16* Bs = smem + 2 * A_HALVES + pa * B_HALVES;
        __syncthreads();
        const u16* asrc = hbf + (size_t)(m0 + ar) * EMB + kp * K_PANEL;
        *reinterpret_cast<uint4*>(&As[ar * LDS_STR + aq * 8]) =
            *reinterpret_cast<const uint4*>(asrc + aq * 8);
        *reinterpret_cast<uint4*>(&As[ar * LDS_STR + (aq + 4) * 8]) =
            *reinterpret_cast<const uint4*>(asrc + (aq + 4) * 8);
        const u16* bsrc = wbf + (size_t)(vb + bc) * EMB + kp * K_PANEL;
#pragma unroll
        for (int i = 0; i < 4; ++i) {
          int k8 = bq + 2 * i;
          *reinterpret_cast<uint4*>(&Bs[bc * LDS_STR + k8 * 8]) =
              *reinterpret_cast<const uint4*>(bsrc + k8 * 8);
        }
        __syncthreads();
      }
#endif
      const u16* Ab = smem + pa * A_HALVES;
      const u16* Bb = smem + 2 * A_HALVES + pa * B_HALVES;
#pragma unroll
      for (int ks = 0; ks < K_PANEL; ks += 32) {
        const int koff = ks + 8 * h;
        v16bf a0 = ld_frag(&Ab[(32 * wm + 0  + ln) * LDS_STR], koff);
        v16bf a1 = ld_frag(&Ab[(32 * wm + 16 + ln) * LDS_STR], koff);
        v16bf b0 = ld_frag(&Bb[(32 * wn + 0  + ln) * LDS_STR], koff);
        v16bf b1 = ld_frag(&Bb[(32 * wn + 16 + ln) * LDS_STR], koff);
        acc[0][0] = __builtin_amdgcn_wmma_f32_16x16x32_bf16(false, a0, false, b0,
                        (short)0, acc[0][0], false, false);
        acc[0][1] = __builtin_amdgcn_wmma_f32_16x16x32_bf16(false, a0, false, b1,
                        (short)0, acc[0][1], false, false);
        acc[1][0] = __builtin_amdgcn_wmma_f32_16x16x32_bf16(false, a1, false, b0,
                        (short)0, acc[1][0], false, false);
        acc[1][1] = __builtin_amdgcn_wmma_f32_16x16x32_bf16(false, a1, false, b1,
                        (short)0, acc[1][1], false, false);
      }
      __syncthreads();   // all waves done with buffer pa before it is re-filled
    }

    // epilogue: bias, target capture, online (max, sum-exp) over 32 cols
    const int c0 = vb + 32 * wn + ln;
    const float b0 = bias[c0];
    const float b1 = bias[c0 + 16];
#pragma unroll
    for (int mt = 0; mt < 2; ++mt) {
#pragma unroll
      for (int j = 0; j < 8; ++j) {
        float a0 = acc[mt][0][j] + b0;   // C layout: VGPR j -> row j+8h, lane -> col
        float a1 = acc[mt][1][j] + b1;
        const int row = m0 + 32 * wm + 16 * mt + j + 8 * h;
        if (tv[mt][j] == c0)      tlogit[row] = a0;   // unique holder: plain store
        if (tv[mt][j] == c0 + 16) tlogit[row] = a1;
        float t = fmaxf(a0, a1);
#pragma unroll
        for (int d = 1; d < 16; d <<= 1) t = fmaxf(t, __shfl_xor(t, d, 16));
        float mn = fmaxf(m_run[mt][j], t);
        float sm = __expf(a0 - mn) + __expf(a1 - mn);
#pragma unroll
        for (int d = 1; d < 16; d <<= 1) sm += __shfl_xor(sm, d, 16);
        s_run[mt][j] = s_run[mt][j] * __expf(m_run[mt][j] - mn) + sm;
        m_run[mt][j] = mn;
      }
    }
  }

  // one lane per half-wave writes its 16 row partials
  if (ln == 0) {
#pragma unroll
    for (int mt = 0; mt < 2; ++mt)
#pragma unroll
      for (int j = 0; j < 8; ++j) {
        int row = m0 + 32 * wm + 16 * mt + j + 8 * h;
        partials[(size_t)row * N_PART + vch * 4 + wn] =
            make_float2(m_run[mt][j], s_run[mt][j]);
      }
  }
}

// ---- kernel 2: per-row combine + block partial reduction -------------------
__global__ __launch_bounds__(256) void flce_rows(
    const float2* __restrict__ partials, const float* __restrict__ tlogit,
    const int* __restrict__ targets, float2* __restrict__ bsum) {
  const int row = blockIdx.x * 256 + threadIdx.x;
  float M = -__builtin_inff(), S = 0.0f;
  const float2* p = partials + (size_t)row * N_PART;
  for (int i = 0; i < N_PART; ++i) {
    float2 v = p[i];
    if (v.x > M) { S = S * __expf(M - v.x) + v.y; M = v.x; }
    else         { S += v.y * __expf(v.x - M); }
  }
  const float lse = M + __logf(S);
  const int t = targets[row];
  const bool valid = (t != -100);
  float nll = valid ? (lse - tlogit[row]) : 0.0f;
  float cnt = valid ? 1.0f : 0.0f;

  __shared__ float sn[256];
  __shared__ float sc[256];
  sn[threadIdx.x] = nll; sc[threadIdx.x] = cnt;
  __syncthreads();
  for (int s = 128; s > 0; s >>= 1) {
    if (threadIdx.x < s) { sn[threadIdx.x] += sn[threadIdx.x + s];
                           sc[threadIdx.x] += sc[threadIdx.x + s]; }
    __syncthreads();
  }
  if (threadIdx.x == 0) bsum[blockIdx.x] = make_float2(sn[0], sc[0]);
}

// ---- kernel 3: deterministic final reduction -------------------------------
__global__ void flce_final(const float2* __restrict__ bsum, float* __restrict__ out) {
  if (threadIdx.x == 0) {
    float s = 0.0f, c = 0.0f;
    for (int i = 0; i < 32; ++i) { s += bsum[i].x; c += bsum[i].y; }
    out[0] = s / fmaxf(c, 1.0f);
  }
}

// ---------------------------------------------------------------------------
extern "C" void kernel_launch(void* const* d_in, const int* in_sizes, int n_in,
                              void* d_out, int out_size, void* d_ws, size_t ws_size,
                              hipStream_t stream) {
  const float* hidden  = (const float*)d_in[0];   // [4,2048,2048]
  const int*   targets = (const int*)d_in[1];     // [8192]
  const float* weight  = (const float*)d_in[2];   // [32000,2048]
  const float* bias    = (const float*)d_in[3];   // [32000]
  float* out = (float*)d_out;

  char* ws = (char*)d_ws;
  u16*    wbf  = (u16*)(ws + OFF_WBF);
  u16*    hbf  = (u16*)(ws + OFF_HBF);
  float*  tlg  = (float*)(ws + OFF_TLG);
  float2* prt  = (float2*)(ws + OFF_PRT);
  float2* bsm  = (float2*)(ws + OFF_BSM);

  // 1) bf16 pre-pass (halves the streamed weight bytes for the 25 V-chunks)
  flce_convert<<<2048, 256, 0, stream>>>(weight, wbf, (size_t)VOCAB * EMB / 8);
  flce_convert<<<2048, 256, 0, stream>>>(hidden, hbf, (size_t)N_ROWS * EMB / 8);
  flce_zero<<<(N_ROWS + 255) / 256, 256, 0, stream>>>(tlg, N_ROWS);

  // 2) fused GEMM + online logsumexp partials
  dim3 grid(N_ROWS / M_BLK, VOCAB / V_CHUNK);   // 128 x 25
  flce_gemm<<<grid, 256, 0, stream>>>(hbf, wbf, bias, targets, tlg, prt);

  // 3) per-row finalize + masked mean
  flce_rows<<<N_ROWS / 256, 256, 0, stream>>>(prt, tlg, targets, bsm);
  flce_final<<<1, 32, 0, stream>>>(bsm, out);
}
